// FraudGNN_27917287424218
// MI455X (gfx1250) — compile-verified, matrix operand order checked by reading, not run on hardware
//
#include <hip/hip_runtime.h>
#include <hip/hip_bf16.h>

// ---------------------------------------------------------------------------
// FraudGNN: GCN(165->128, ReLU) -> GAT(128->64, softmax attn, ReLU) -> FC(64->2)
// N=100000 nodes, E=1600000 edges. Memory/atomic bound; GEMMs use WMMA f16.
// ---------------------------------------------------------------------------

constexpr int IN_DIM = 165;
constexpr int HID    = 128;
constexpr int OUT_H  = 64;

typedef __attribute__((ext_vector_type(16))) _Float16 v16h;
typedef __attribute__((ext_vector_type(8)))  float    v8f;

// ---- monotonic float <-> uint key for atomic max over signed floats --------
__device__ __forceinline__ unsigned ordKey(float f) {
    unsigned u = __float_as_uint(f);
    return (u & 0x80000000u) ? ~u : (u | 0x80000000u);
}
__device__ __forceinline__ float ordInv(unsigned k) {
    return __uint_as_float((k & 0x80000000u) ? (k & 0x7fffffffu) : ~k);
}
__device__ __forceinline__ float leaky02(float v) { return v > 0.0f ? v : 0.2f * v; }

// ---------------------------------------------------------------------------
// WMMA GEMM: C[M x NC] = A[M x K] * B[K x NC], f32 in/out, f16 multiply,
// f32 accumulate via v_wmma_f32_16x16x32_f16.
//  - LDS tiles stored directly in WMMA fragment layout (32B/lane) so the
//    fragment read is one v16h per operand (ds_load_b128 x2).
//  - Staging is two-phase (unrolled batched global loads into registers, then
//    unrolled LDS stores) so the backend issues all loads before one wait.
//  - All bounds handling is branch-free: row/K indices clamped to stay in the
//    tensor, out-of-range VALUES zeroed by select (no exec-mask serialization).
//  - RELU_A fuses relu into A staging; optional fused epilogue computes
//    Hout = bias + C * dinv(m)^2  (GCN self-loop init).
// Fragment map (ISA 7.12.2, 16-bit 16x32 A / 32x16 B):
//   laneT = (r&15) | ((kk&8)<<1) ;  j = (kk&7) | ((kk&16)>>1)
// ---------------------------------------------------------------------------
template <int RW, int CW, bool RELU_A>
__global__ void wmma_gemm_kernel(const float* __restrict__ A,
                                 const float* __restrict__ B,
                                 float* __restrict__ C,
                                 const float* __restrict__ bias,
                                 const float* __restrict__ dinv,
                                 float* __restrict__ Hout,
                                 int M, int K, int NC) {
    constexpr int ROWS    = RW * 16;
    constexpr int COLS    = CW * 16;
    constexpr int NTHR    = RW * CW * 32;
    constexpr int A_ITERS = (ROWS * 32) / NTHR;   // 2 (1x8) or 4 (2x4)
    constexpr int B_ITERS = (32 * COLS) / NTHR;   // 16 (1x8) or 8 (2x4)

    __shared__ __align__(32) _Float16 sAf[RW][32][16];   // fragment-layout A
    __shared__ __align__(32) _Float16 sBf[CW][32][16];   // fragment-layout B

    const int tid      = threadIdx.x;
    const int lane     = tid & 31;
    const int wave     = tid >> 5;
    const int waveRow  = wave / CW;
    const int waveCol  = wave % CW;
    const int hi       = lane >> 4;
    const int l16      = lane & 15;
    const int blockRow = blockIdx.x * ROWS;

    v8f acc = {};

    // stage one 32-deep K chunk into LDS (fragment layout), branch-free.
    // tail==true additionally zeroes K-overhang values (addresses stay valid
    // via clamping, so loads are never exec-masked).
    auto stage = [&](int k0, bool tail) {
        float va[A_ITERS];
        float vb[B_ITERS];
#pragma unroll
        for (int it = 0; it < A_ITERS; ++it) {
            int idx = tid + it * NTHR;
            int r = idx >> 5, kk = idx & 31;
            int gm = blockRow + r;
            gm = gm < M ? gm : M - 1;                 // row clamp (row-local GEMM)
            int gk = k0 + kk;
            int gkc = (!tail || gk < K) ? gk : K - 1; // K clamp (valid address)
            const float* gp = A + (long)gm * K + gkc;
            float v = *gp;
            if (!tail) __builtin_prefetch(gp + 32, 0, 0);   // next K chunk of A
            va[it] = (!tail || gk < K) ? v : 0.0f;    // select, not exec-mask
        }
#pragma unroll
        for (int it = 0; it < B_ITERS; ++it) {
            int idx = tid + it * NTHR;
            int kk = idx / COLS, c = idx % COLS;
            int gk = k0 + kk;
            int gkc = (!tail || gk < K) ? gk : K - 1;
            float v = B[(long)gkc * NC + c];
            vb[it] = (!tail || gk < K) ? v : 0.0f;
        }
#pragma unroll
        for (int it = 0; it < A_ITERS; ++it) {
            int idx = tid + it * NTHR;
            int r = idx >> 5, kk = idx & 31;
            float v = RELU_A ? fmaxf(va[it], 0.0f) : va[it];
            sAf[r >> 4][(r & 15) | ((kk & 8) << 1)][(kk & 7) | ((kk & 16) >> 1)] =
                (_Float16)v;
        }
#pragma unroll
        for (int it = 0; it < B_ITERS; ++it) {
            int idx = tid + it * NTHR;
            int kk = idx / COLS, c = idx % COLS;
            sBf[c >> 4][(c & 15) | ((kk & 8) << 1)][(kk & 7) | ((kk & 16) >> 1)] =
                (_Float16)vb[it];
        }
    };
    auto mma = [&]() {
        const v16h af = *(const v16h*)sAf[waveRow][lane];
        const v16h bf = *(const v16h*)sBf[waveCol][lane];
        acc = __builtin_amdgcn_wmma_f32_16x16x32_f16(
            /*neg_a=*/false, af, /*neg_b=*/false, bf,
            /*c_mod=*/(short)0, acc, /*reuse_a=*/false, /*reuse_b=*/false);
    };

    const int kfull = K >> 5;
    for (int kc = 0; kc < kfull; ++kc) {
        stage(kc * 32, false);
        __syncthreads();
        mma();
        __syncthreads();
    }
    if (K & 31) {
        stage(kfull * 32, true);
        __syncthreads();
        mma();
        __syncthreads();
    }

    // epilogue: C/D layout -> VGPR r holds M=r (lanes 0-15) / M=r+8 (16-31)
#pragma unroll
    for (int r = 0; r < 8; ++r) {
        int m = blockRow + waveRow * 16 + r + hi * 8;
        int c = waveCol * 16 + l16;
        if (m < M) {
            float v = acc[r];
            C[(long)m * NC + c] = v;
            if (Hout) {                                  // fused GCN init
                float d = dinv[m];
                Hout[(long)m * NC + c] = bias[c] + v * d * d;
            }
        }
    }
}

// ------------------------- elementwise / edge kernels -----------------------
__global__ void k_fill1(float* p, int n) {
    int i = blockIdx.x * blockDim.x + threadIdx.x;
    if (i < n) p[i] = 1.0f;                 // self-loop degree
}
__global__ void k_deg(const int* __restrict__ col, float* deg, int E) {
    int e = blockIdx.x * blockDim.x + threadIdx.x;
    if (e < E) atomicAdd(&deg[col[e]], 1.0f);
}
__global__ void k_rsqrt(float* p, int n) {
    int i = blockIdx.x * blockDim.x + threadIdx.x;
    if (i < n) p[i] = rsqrtf(p[i]);
}
// one wave per edge: gather T1[row], scale, atomic-scatter into H1[col]
__global__ void k_gcn_scatter(const int* __restrict__ row, const int* __restrict__ col,
                              const float* __restrict__ T1, const float* __restrict__ dinv,
                              float* __restrict__ H1, int E) {
    int e    = (blockIdx.x * blockDim.x + threadIdx.x) >> 5;
    int lane = threadIdx.x & 31;
    if (e >= E) return;
    int r = row[e], c = col[e];
    float w = dinv[r] * dinv[c];
    const float4 t = *(const float4*)(T1 + (long)r * HID + lane * 4);
    float* dst = H1 + (long)c * HID + lane * 4;
    atomicAdd(dst + 0, t.x * w);
    atomicAdd(dst + 1, t.y * w);
    atomicAdd(dst + 2, t.z * w);
    atomicAdd(dst + 3, t.w * w);
}
// per-node attention scores es=G.a_src, ed=G.a_dst; init segment max w/ self edge
__global__ void k_scores(const float* __restrict__ G, const float* __restrict__ a_src,
                         const float* __restrict__ a_dst, float* __restrict__ es,
                         float* __restrict__ ed, unsigned* __restrict__ mmax, int n) {
    int i    = (blockIdx.x * blockDim.x + threadIdx.x) >> 5;
    int lane = threadIdx.x & 31;
    if (i >= n) return;
    float g0 = G[(long)i * OUT_H + lane];
    float g1 = G[(long)i * OUT_H + 32 + lane];
    float ps = g0 * a_src[lane] + g1 * a_src[32 + lane];
    float pd = g0 * a_dst[lane] + g1 * a_dst[32 + lane];
#pragma unroll
    for (int o = 16; o > 0; o >>= 1) {
        ps += __shfl_xor(ps, o, 32);
        pd += __shfl_xor(pd, o, 32);
    }
    if (lane == 0) {
        es[i] = ps;
        ed[i] = pd;
        mmax[i] = ordKey(leaky02(ps + pd));   // self loop seeds the max
    }
}
__global__ void k_att_max(const int* __restrict__ row, const int* __restrict__ col,
                          const float* __restrict__ es, const float* __restrict__ ed,
                          unsigned* __restrict__ mmax, int E) {
    int e = blockIdx.x * blockDim.x + threadIdx.x;
    if (e >= E) return;
    float v = leaky02(es[row[e]] + ed[col[e]]);
    atomicMax(&mmax[col[e]], ordKey(v));
}
// seed softmax denominator with self-loop exp
__global__ void k_att_self(const float* __restrict__ es, const float* __restrict__ ed,
                           const unsigned* __restrict__ mmax, float* __restrict__ ssum, int n) {
    int i = blockIdx.x * blockDim.x + threadIdx.x;
    if (i < n) ssum[i] = expf(leaky02(es[i] + ed[i]) - ordInv(mmax[i]));
}
// seed numerator with self-loop term
__global__ void k_num_init(const float* __restrict__ G, const float* __restrict__ es,
                           const float* __restrict__ ed, const unsigned* __restrict__ mmax,
                           float* __restrict__ num, int total) {
    int idx = blockIdx.x * blockDim.x + threadIdx.x;
    if (idx < total) {
        int i = idx >> 6;   // /OUT_H
        float ex = expf(leaky02(es[i] + ed[i]) - ordInv(mmax[i]));
        num[idx] = ex * G[idx];
    }
}
// one wave per edge: accumulate softmax denom + weighted numerator
__global__ void k_att_edges(const int* __restrict__ row, const int* __restrict__ col,
                            const float* __restrict__ es, const float* __restrict__ ed,
                            const unsigned* __restrict__ mmax, float* __restrict__ ssum,
                            const float* __restrict__ G, float* __restrict__ num, int E) {
    int e    = (blockIdx.x * blockDim.x + threadIdx.x) >> 5;
    int lane = threadIdx.x & 31;
    if (e >= E) return;
    int r = row[e], c = col[e];
    float ex = expf(leaky02(es[r] + ed[c]) - ordInv(mmax[c]));
    if (lane == 0) atomicAdd(&ssum[c], ex);
    float g0 = G[(long)r * OUT_H + lane];
    float g1 = G[(long)r * OUT_H + 32 + lane];
    atomicAdd(&num[(long)c * OUT_H + lane],      ex * g0);
    atomicAdd(&num[(long)c * OUT_H + 32 + lane], ex * g1);
}
// finalize: h = relu(num/s + bg); out = h @ Wfc + bfc   (one wave per node)
__global__ void k_final(const float* __restrict__ num, const float* __restrict__ ssum,
                        const float* __restrict__ bg, const float* __restrict__ Wfc,
                        const float* __restrict__ bfc, float* __restrict__ out, int n) {
    int i    = (blockIdx.x * blockDim.x + threadIdx.x) >> 5;
    int lane = threadIdx.x & 31;
    if (i >= n) return;
    float sinv = 1.0f / ssum[i];
    int f0 = lane, f1 = lane + 32;
    float h0 = fmaxf(num[(long)i * OUT_H + f0] * sinv + bg[f0], 0.0f);
    float h1 = fmaxf(num[(long)i * OUT_H + f1] * sinv + bg[f1], 0.0f);
    float a0 = h0 * Wfc[f0 * 2 + 0] + h1 * Wfc[f1 * 2 + 0];
    float a1 = h0 * Wfc[f0 * 2 + 1] + h1 * Wfc[f1 * 2 + 1];
#pragma unroll
    for (int o = 16; o > 0; o >>= 1) {
        a0 += __shfl_xor(a0, o, 32);
        a1 += __shfl_xor(a1, o, 32);
    }
    if (lane == 0) {
        out[i * 2 + 0] = a0 + bfc[0];
        out[i * 2 + 1] = a1 + bfc[1];
    }
}

// ---------------------------------------------------------------------------
extern "C" void kernel_launch(void* const* d_in, const int* in_sizes, int n_in,
                              void* d_out, int out_size, void* d_ws, size_t ws_size,
                              hipStream_t stream) {
    const float* x     = (const float*)d_in[0];
    const int*   eidx  = (const int*)d_in[1];   // [2, E] int32
    const float* W1    = (const float*)d_in[2];
    const float* b1    = (const float*)d_in[3];
    const float* Wg    = (const float*)d_in[4];
    const float* a_src = (const float*)d_in[5];
    const float* a_dst = (const float*)d_in[6];
    const float* bg    = (const float*)d_in[7];
    const float* Wfc   = (const float*)d_in[8];
    const float* bfc   = (const float*)d_in[9];
    float* out = (float*)d_out;

    const int n = in_sizes[0] / IN_DIM;
    const int E = in_sizes[1] / 2;
    const int* row = eidx;
    const int* col = eidx + E;

    // workspace layout (floats): dinv | es | ed | mmax | ssum | bufA(N*128) | bufB(N*128)
    float* p = (float*)d_ws;
    float*    dinv = p;              p += n;
    float*    es   = p;              p += n;
    float*    ed   = p;              p += n;
    unsigned* mmax = (unsigned*)p;   p += n;
    float*    ssum = p;              p += n;
    float*    bufA = p;              p += (size_t)n * HID;
    float*    bufB = p;
    float* T1  = bufA;   // X@W1
    float* H1  = bufB;   // GCN pre-activation (relu fused into GEMM2 staging)
    float* G   = bufA;   // relu(H1)@Wg (reuses T1 after scatter done)
    float* num = bufB;   // GAT numerator (reuses H1 after GEMM2 done)

    const int T = 256;
    const int tot2 = n * OUT_H;

    // --- GCN layer ---
    k_fill1 <<<(n + T - 1) / T, T, 0, stream>>>(dinv, n);
    k_deg   <<<(E + T - 1) / T, T, 0, stream>>>(col, dinv, E);
    k_rsqrt <<<(n + T - 1) / T, T, 0, stream>>>(dinv, n);
    // T1 = x@W1 ; fused: H1 = b1 + T1*dinv^2  (self-loop message + bias)
    wmma_gemm_kernel<1, 8, false><<<(n + 15) / 16, 256, 0, stream>>>(
        x, W1, T1, b1, dinv, H1, n, IN_DIM, HID);
    k_gcn_scatter<<<(E + 7) / 8, 256, 0, stream>>>(row, col, T1, dinv, H1, E);

    // --- GAT layer ---  (relu of H1 fused into A-staging)
    wmma_gemm_kernel<2, 4, true><<<(n + 31) / 32, 256, 0, stream>>>(
        H1, Wg, G, nullptr, nullptr, nullptr, n, HID, OUT_H);
    k_scores   <<<(n + 7) / 8, 256, 0, stream>>>(G, a_src, a_dst, es, ed, mmax, n);
    k_att_max  <<<(E + T - 1) / T, T, 0, stream>>>(row, col, es, ed, mmax, E);
    k_att_self <<<(n + T - 1) / T, T, 0, stream>>>(es, ed, mmax, ssum, n);
    k_num_init <<<(tot2 + T - 1) / T, T, 0, stream>>>(G, es, ed, mmax, num, tot2);
    k_att_edges<<<(E + 7) / 8, 256, 0, stream>>>(row, col, es, ed, mmax, ssum, G, num, E);

    // --- final FC ---
    k_final<<<(n + 7) / 8, 256, 0, stream>>>(num, ssum, bg, Wfc, bfc, out, n);

    (void)n_in; (void)out_size; (void)ws_size;
}
